// MatchLSTM_85744727097434
// MI455X (gfx1250) — compile-verified, matrix operand order checked by reading, not run on hardware
//
#include <hip/hip_runtime.h>
#include <hip/hip_bf16.h>
#include <cstdint>

// ---------------------------------------------------------------------------
// MatchLSTM on gfx1250: bf16 WMMA GEMMs (v_wmma_f32_16x16x32_bf16) with f32
// accumulation, f32 cell state / softmax. K dims padded to multiples of 32
// (300->320, 600->640); gate axis padded 1200->1216 so each wave can compute
// a 2x2 block of 16x16 tiles (32x32 of C) -> A/B fragment reuse halves the
// load:WMMA ratio and 4 independent accumulators give WMMA-level ILP.
// ---------------------------------------------------------------------------

typedef __bf16 bf16_t;
typedef bf16_t v16bf __attribute__((ext_vector_type(16)));
typedef bf16_t v8bf  __attribute__((ext_vector_type(8)));
typedef float  v8f   __attribute__((ext_vector_type(8)));

#define DEV __device__ __forceinline__

DEV unsigned short f2bf_bits(float f) {
    union { float f; unsigned u; } v; v.f = f;
    unsigned u = v.u;
    u += 0x7FFFu + ((u >> 16) & 1u);   // round-to-nearest-even
    return (unsigned short)(u >> 16);
}
DEV float bf2f_bits(unsigned short b) {
    union { unsigned u; float f; } v; v.u = ((unsigned)b) << 16;
    return v.f;
}
DEV float sigmoidf_(float x) { return 1.0f / (1.0f + expf(-x)); }

// ---------------------------------------------------------------------------
__global__ void k_zero(uint32_t* p, size_t nwords) {
    size_t i = (size_t)blockIdx.x * blockDim.x + threadIdx.x;
    size_t stride = (size_t)gridDim.x * blockDim.x;
    for (; i < nwords; i += stride) p[i] = 0u;
}

// fp32 [rows x cols] -> bf16 [rowsOut x colsOut], zero padded
__global__ void k_cvt_pad(const float* __restrict__ src, unsigned short* __restrict__ dst,
                          int rows, int cols, int rowsOut, int colsOut) {
    int i = blockIdx.x * blockDim.x + threadIdx.x;
    int total = rowsOut * colsOut;
    if (i >= total) return;
    int r = i / colsOut, c = i - r * colsOut;
    unsigned short v = 0;
    if (r < rows && c < cols) v = f2bf_bits(src[(size_t)r * cols + c]);
    dst[i] = v;
}

__global__ void k_bias_sum(const float* __restrict__ a, const float* __restrict__ b,
                           float* __restrict__ o, int n) {
    int i = blockIdx.x * blockDim.x + threadIdx.x;
    if (i < n) o[i] = a[i] + b[i];
}

// gather embeddings -> bf16, padded row stride Epad (pads pre-zeroed)
__global__ void k_embed(const int* __restrict__ tok, const float* __restrict__ emb,
                        unsigned short* __restrict__ dst, int T, int Bb, int E, int Epad) {
    int i = blockIdx.x * blockDim.x + threadIdx.x;
    int total = T * Bb * E;
    if (i >= total) return;
    int e = i % E;
    int tb = i / E;                    // tb = t*B + b, matches [T,B] token layout
    int token = tok[tb];
    dst[(size_t)tb * Epad + e] = f2bf_bits(emb[(size_t)token * E + e]);
}

// ---------------------------------------------------------------------------
// C[M x N] = A0[M x K]·W0[N x K]^T (+ A1[M x K]·W1[N x K]^T) (+ bias[N])
// One wave computes a TMB x TNB block of 16x16 tiles. K multiple of 32,
// M multiple of 16*TMB, N multiple of 16*TNB.
// Fragment layouts per CDNA5 ISA 7.12.2 (16-bit A 16x32 / B 32x16 / f32 C 16x16).
// ---------------------------------------------------------------------------
template<int TMB, int TNB>
__global__ void __launch_bounds__(256)
k_wmma_gemm(const unsigned short* __restrict__ A0, const unsigned short* __restrict__ W0,
            const unsigned short* __restrict__ A1, const unsigned short* __restrict__ W1,
            const float* __restrict__ bias, float* __restrict__ C,
            int M, int N, int K, int lda, int ldw, int ldc)
{
    int wave = threadIdx.x >> 5;
    int lane = threadIdx.x & 31;
    int blk  = blockIdx.x * (blockDim.x >> 5) + wave;
    int nbN  = N / (16 * TNB);
    int nblk = (M / (16 * TMB)) * nbN;
    if (blk >= nblk) return;                    // wave-uniform: EXEC all-ones inside

    int bm = blk / nbN, bn = blk - bm * nbN;
    int half = lane >> 4;                       // 0: k 0-7 and 16-23, 1: k 8-15 and 24-31
    int l15  = lane & 15;
    int kgrp = half * 8;

    v8f acc[TMB][TNB];
#pragma unroll
    for (int i = 0; i < TMB; ++i)
#pragma unroll
        for (int j = 0; j < TNB; ++j) acc[i][j] = {};

    for (int pass = 0; pass < 2; ++pass) {
        const unsigned short* Ab = (pass == 0) ? A0 : A1;
        const unsigned short* Wb = (pass == 0) ? W0 : W1;
        if (!Ab) break;                         // uniform across wave

        const unsigned short* ap[TMB];
        const unsigned short* wp[TNB];
#pragma unroll
        for (int i = 0; i < TMB; ++i)
            ap[i] = Ab + (size_t)(bm * 16 * TMB + i * 16 + l15) * lda + kgrp;
#pragma unroll
        for (int j = 0; j < TNB; ++j)
            wp[j] = Wb + (size_t)(bn * 16 * TNB + j * 16 + l15) * ldw + kgrp;

        for (int k = 0; k < K; k += 32) {
            v16bf af[TMB], bf[TNB];
#pragma unroll
            for (int i = 0; i < TMB; ++i) {
                v8bf lo = *reinterpret_cast<const v8bf*>(ap[i] + k);
                v8bf hi = *reinterpret_cast<const v8bf*>(ap[i] + k + 16);
#pragma unroll
                for (int x = 0; x < 8; ++x) { af[i][x] = lo[x]; af[i][x + 8] = hi[x]; }
            }
#pragma unroll
            for (int j = 0; j < TNB; ++j) {
                v8bf lo = *reinterpret_cast<const v8bf*>(wp[j] + k);
                v8bf hi = *reinterpret_cast<const v8bf*>(wp[j] + k + 16);
#pragma unroll
                for (int x = 0; x < 8; ++x) { bf[j][x] = lo[x]; bf[j][x + 8] = hi[x]; }
            }
#pragma unroll
            for (int i = 0; i < TMB; ++i)
#pragma unroll
                for (int j = 0; j < TNB; ++j)
                    acc[i][j] = __builtin_amdgcn_wmma_f32_16x16x32_bf16(
                        false, af[i], false, bf[j], (short)0, acc[i][j], false, false);
        }
    }

#pragma unroll
    for (int i = 0; i < TMB; ++i) {
#pragma unroll
        for (int j = 0; j < TNB; ++j) {
            int c = bn * 16 * TNB + j * 16 + l15;
            float bv = bias ? bias[c] : 0.0f;
#pragma unroll
            for (int v = 0; v < 8; ++v) {       // VGPR v -> row v (+8 for upper lanes)
                int r = bm * 16 * TMB + i * 16 + v + half * 8;
                C[(size_t)r * ldc + c] = acc[i][j][v] + bv;
            }
        }
    }
}

// ---------------------------------------------------------------------------
// LSTM nonlinearity: gates [B x ldg] (i,f,g,o at 0,H,2H,3H) -> c', h
__global__ void k_lstm_pointwise(const float* __restrict__ gates,
                                 const float* __restrict__ c_in,
                                 float* __restrict__ c_out,
                                 float* __restrict__ h_f32,
                                 unsigned short* __restrict__ h_bf,
                                 int Bb, int H, int Hpad, int ldg)
{
    int i = blockIdx.x * blockDim.x + threadIdx.x;
    if (i >= Bb * H) return;
    int b = i / H, h = i - b * H;
    const float* g = gates + (size_t)b * ldg;
    float gi = g[h], gf = g[H + h], gg = g[2 * H + h], go = g[3 * H + h];
    float c2 = sigmoidf_(gf) * c_in[i] + sigmoidf_(gi) * tanhf(gg);
    float hv = sigmoidf_(go) * tanhf(c2);
    c_out[i] = c2;
    h_f32[i] = hv;
    h_bf[(size_t)b * Hpad + h] = f2bf_bits(hv);
}

// e[t,b] = sum_h tanh(ws_hs[t,b,h] + q[b,h]) * w_e[h]   (one wave per (t,b))
__global__ void __launch_bounds__(256)
k_attn_e(const float* __restrict__ ws_hs, const float* __restrict__ q,
         const float* __restrict__ w_e, float* __restrict__ e,
         int TPb, int Bb, int H, int Hpad)
{
    int wave = threadIdx.x >> 5, lane = threadIdx.x & 31;
    int pair = blockIdx.x * (blockDim.x >> 5) + wave;
    if (pair >= TPb * Bb) return;
    int b = pair % Bb;
    const float* ws = ws_hs + (size_t)pair * Hpad;
    const float* qb = q + (size_t)b * Hpad;
    float acc = 0.f;
    for (int h = lane; h < H; h += 32)
        acc += tanhf(ws[h] + qb[h]) * w_e[h];
    for (int off = 16; off; off >>= 1) acc += __shfl_down(acc, off, 32);
    if (lane == 0) e[pair] = acc;
}

// softmax over t (TPb<=128, multiple of 32); one wave per batch element
__global__ void __launch_bounds__(256)
k_softmax_t(const float* __restrict__ e, float* __restrict__ alpha, int TPb, int Bb)
{
    int wave = threadIdx.x >> 5, lane = threadIdx.x & 31;
    int b = blockIdx.x * (blockDim.x >> 5) + wave;
    if (b >= Bb) return;
    int nper = TPb >> 5;             // 4 for TP=128
    float v[4];
    float mx = -1e30f;
    for (int j = 0; j < nper; ++j) { v[j] = e[(j * 32 + lane) * Bb + b]; mx = fmaxf(mx, v[j]); }
    for (int off = 16; off; off >>= 1) mx = fmaxf(mx, __shfl_xor(mx, off, 32));
    float s = 0.f;
    for (int j = 0; j < nper; ++j) { v[j] = expf(v[j] - mx); s += v[j]; }
    for (int off = 16; off; off >>= 1) s += __shfl_xor(s, off, 32);
    float inv = 1.0f / s;
    for (int j = 0; j < nper; ++j) alpha[(j * 32 + lane) * Bb + b] = v[j] * inv;
}

// a[b,h] = sum_t alpha[t,b]*h_s[t,b,h]; writes m = [a, h_t] in padded bf16
__global__ void k_attn_combine(const float* __restrict__ alpha,
                               const unsigned short* __restrict__ hs_bf,
                               const float* __restrict__ ht_f32,
                               unsigned short* __restrict__ m_bf,
                               int TPb, int Bb, int H, int Hpad, int Mld)
{
    int i = blockIdx.x * blockDim.x + threadIdx.x;
    if (i >= Bb * H) return;
    int b = i / H, h = i - b * H;
    float acc = 0.f;
    for (int t = 0; t < TPb; ++t)
        acc += alpha[t * Bb + b] * bf2f_bits(hs_bf[((size_t)t * Bb + b) * Hpad + h]);
    m_bf[(size_t)b * Mld + h]     = f2bf_bits(acc);
    m_bf[(size_t)b * Mld + H + h] = f2bf_bits(ht_f32[i]);
}

// out[b,c] = sum_h hm[b,h]*fc_w[c,h] + fc_b[c]   (one wave per (b,c))
__global__ void __launch_bounds__(256)
k_fc(const float* __restrict__ hm, const float* __restrict__ w,
     const float* __restrict__ bias, float* __restrict__ out,
     int Bb, int H, int Cc)
{
    int wave = threadIdx.x >> 5, lane = threadIdx.x & 31;
    int idx = blockIdx.x * (blockDim.x >> 5) + wave;
    if (idx >= Bb * Cc) return;
    int b = idx / Cc, c = idx - b * Cc;
    float acc = 0.f;
    for (int h = lane; h < H; h += 32) acc += hm[(size_t)b * H + h] * w[(size_t)c * H + h];
    for (int off = 16; off; off >>= 1) acc += __shfl_down(acc, off, 32);
    if (lane == 0) out[idx] = acc + bias[c];
}

// ---------------------------------------------------------------------------
extern "C" void kernel_launch(void* const* d_in, const int* in_sizes, int n_in,
                              void* d_out, int out_size, void* d_ws, size_t ws_size,
                              hipStream_t stream)
{
    (void)in_sizes; (void)n_in; (void)out_size;
    const int TP = 128, TH = 64, Bb = 256, E = 300, H = 300, Cc = 3;
    const int Epad = 320, Hpad = 320, G = 1200, Gp = 1216, Mcat = 600, Mpad = 640;

    const int*   premise  = (const int*)d_in[0];
    const int*   hypoth   = (const int*)d_in[2];
    const float* word2vec = (const float*)d_in[4];
    const float* w_e      = (const float*)d_in[5];
    const float* W_s      = (const float*)d_in[6];
    const float* W_t      = (const float*)d_in[7];
    const float* W_m      = (const float*)d_in[8];
    const float* fc_w     = (const float*)d_in[9];
    const float* fc_b     = (const float*)d_in[10];
    const float* pWih = (const float*)d_in[11];
    const float* pWhh = (const float*)d_in[12];
    const float* pbih = (const float*)d_in[13];
    const float* pbhh = (const float*)d_in[14];
    const float* hWih = (const float*)d_in[15];
    const float* hWhh = (const float*)d_in[16];
    const float* hbih = (const float*)d_in[17];
    const float* hbhh = (const float*)d_in[18];
    const float* mWih = (const float*)d_in[19];
    const float* mbih = (const float*)d_in[21];
    const float* mbhh = (const float*)d_in[22];
    float* out = (float*)d_out;

    char* ws = (char*)d_ws;
    size_t off = 0;
    auto alloc = [&](size_t bytes) -> char* {
        char* p = ws + off;
        off += (bytes + 255) & ~(size_t)255;
        return p;
    };

    unsigned short* prem_x   = (unsigned short*)alloc((size_t)TP * Bb * Epad * 2);
    unsigned short* hypo_x   = (unsigned short*)alloc((size_t)TH * Bb * Epad * 2);
    unsigned short* hs_bf    = (unsigned short*)alloc((size_t)TP * Bb * Hpad * 2);
    float*          ws_hs_f  = (float*)alloc((size_t)TP * Bb * Hpad * 4);
    unsigned short* pWih_bf  = (unsigned short*)alloc((size_t)Gp * Epad * 2);
    unsigned short* pWhh_bf  = (unsigned short*)alloc((size_t)Gp * Hpad * 2);
    unsigned short* hWih_bf  = (unsigned short*)alloc((size_t)Gp * Epad * 2);
    unsigned short* hWhh_bf  = (unsigned short*)alloc((size_t)Gp * Hpad * 2);
    unsigned short* mWih_bf  = (unsigned short*)alloc((size_t)Gp * Mpad * 2);
    unsigned short* Ws_bf    = (unsigned short*)alloc((size_t)Hpad * Hpad * 2);
    unsigned short* Wt_bf    = (unsigned short*)alloc((size_t)Hpad * Hpad * 2);
    unsigned short* Wm_bf    = (unsigned short*)alloc((size_t)Hpad * Hpad * 2);
    float*          bias_p   = (float*)alloc((size_t)Gp * 4);
    float*          bias_h   = (float*)alloc((size_t)Gp * 4);
    float*          bias_m   = (float*)alloc((size_t)Gp * 4);
    float*          gates    = (float*)alloc((size_t)Bb * Gp * 4);
    float*          c_prem   = (float*)alloc((size_t)Bb * H * 4);
    float*          c_hyp    = (float*)alloc((size_t)Bb * H * 4);
    float*          c_dummy  = (float*)alloc((size_t)Bb * H * 4);
    float*          h_sink   = (float*)alloc((size_t)Bb * H * 4);
    float*          ht_f32   = (float*)alloc((size_t)Bb * H * 4);
    float*          hm_f32   = (float*)alloc((size_t)Bb * H * 4);
    float*          zero_f32 = (float*)alloc((size_t)Bb * H * 4);
    unsigned short* ht_bf    = (unsigned short*)alloc((size_t)Bb * Hpad * 2);
    unsigned short* hm_bf    = (unsigned short*)alloc((size_t)Bb * Hpad * 2);
    unsigned short* zero_bf  = (unsigned short*)alloc((size_t)Bb * Hpad * 2);
    float*          q_f      = (float*)alloc((size_t)Bb * Hpad * 4);
    float*          e_buf    = (float*)alloc((size_t)TP * Bb * 4);
    float*          alpha_f  = (float*)alloc((size_t)TP * Bb * 4);
    unsigned short* m_bf     = (unsigned short*)alloc((size_t)Bb * Mpad * 2);

    size_t total = off;
    if (ws_size < total) return;   // deterministic no-op fallback

    // --- setup: zero workspace (pads/initial states), convert weights ---
    k_zero<<<4096, 256, 0, stream>>>((uint32_t*)ws, total / 4);

    auto cvt = [&](const float* src, unsigned short* dst, int r, int c, int ro, int co) {
        int n = ro * co;
        k_cvt_pad<<<(n + 255) / 256, 256, 0, stream>>>(src, dst, r, c, ro, co);
    };
    cvt(pWih, pWih_bf, G, E, Gp, Epad);
    cvt(pWhh, pWhh_bf, G, H, Gp, Hpad);
    cvt(hWih, hWih_bf, G, E, Gp, Epad);
    cvt(hWhh, hWhh_bf, G, H, Gp, Hpad);
    cvt(mWih, mWih_bf, G, Mcat, Gp, Mpad);
    cvt(W_s, Ws_bf, H, H, Hpad, Hpad);
    cvt(W_t, Wt_bf, H, H, Hpad, Hpad);
    cvt(W_m, Wm_bf, H, H, Hpad, Hpad);

    k_bias_sum<<<(G + 255) / 256, 256, 0, stream>>>(pbih, pbhh, bias_p, G);
    k_bias_sum<<<(G + 255) / 256, 256, 0, stream>>>(hbih, hbhh, bias_h, G);
    k_bias_sum<<<(G + 255) / 256, 256, 0, stream>>>(mbih, mbhh, bias_m, G);

    k_embed<<<(TP * Bb * E + 255) / 256, 256, 0, stream>>>(premise, word2vec, prem_x, TP, Bb, E, Epad);
    k_embed<<<(TH * Bb * E + 255) / 256, 256, 0, stream>>>(hypoth, word2vec, hypo_x, TH, Bb, E, Epad);

    // 2x2 tile-blocked WMMA GEMM launcher: M%32==0, N%32==0, K%32==0
    auto gemm = [&](const unsigned short* A0, const unsigned short* W0,
                    const unsigned short* A1, const unsigned short* W1,
                    const float* bias, float* Cp,
                    int M, int N, int K, int lda, int ldw, int ldc) {
        int nblk = (M / 32) * (N / 32);
        k_wmma_gemm<2, 2><<<(nblk + 7) / 8, 256, 0, stream>>>(A0, W0, A1, W1, bias, Cp,
                                                              M, N, K, lda, ldw, ldc);
    };
    const int nBH = Bb * H;

    // --- premise LSTM (128 sequential steps) ---
    const unsigned short* hprev = zero_bf;
    const float* cprev = zero_f32;
    for (int t = 0; t < TP; ++t) {
        gemm(prem_x + (size_t)t * Bb * Epad, pWih_bf, hprev, pWhh_bf, bias_p, gates,
             Bb, Gp, Epad, Epad, Epad, Gp);
        unsigned short* hs_t = hs_bf + (size_t)t * Bb * Hpad;
        k_lstm_pointwise<<<(nBH + 255) / 256, 256, 0, stream>>>(gates, cprev, c_prem,
                                                                h_sink, hs_t, Bb, H, Hpad, Gp);
        hprev = hs_t; cprev = c_prem;
    }

    // --- ws_hs = h_s @ W_s^T  (one big 32768 x 320 x 320 GEMM) ---
    gemm(hs_bf, Ws_bf, nullptr, nullptr, nullptr, ws_hs_f,
         TP * Bb, Hpad, Hpad, Hpad, Hpad, Hpad);

    // --- hypothesis loop (64 sequential steps) ---
    for (int t = 0; t < TH; ++t) {
        // hypo LSTM cell
        gemm(hypo_x + (size_t)t * Bb * Epad, hWih_bf, ht_bf, hWhh_bf, bias_h, gates,
             Bb, Gp, Epad, Epad, Epad, Gp);
        k_lstm_pointwise<<<(nBH + 255) / 256, 256, 0, stream>>>(gates, c_hyp, c_hyp,
                                                                ht_f32, ht_bf, Bb, H, Hpad, Gp);
        // q = h_t @ W_t^T + h_m @ W_m^T
        gemm(ht_bf, Wt_bf, hm_bf, Wm_bf, nullptr, q_f,
             Bb, Hpad, Hpad, Hpad, Hpad, Hpad);
        // attention over premise positions
        k_attn_e<<<(TP * Bb + 7) / 8, 256, 0, stream>>>(ws_hs_f, q_f, w_e, e_buf, TP, Bb, H, Hpad);
        k_softmax_t<<<(Bb + 7) / 8, 256, 0, stream>>>(e_buf, alpha_f, TP, Bb);
        k_attn_combine<<<(nBH + 255) / 256, 256, 0, stream>>>(alpha_f, hs_bf, ht_f32, m_bf,
                                                              TP, Bb, H, Hpad, Mpad);
        // match cell with zero hx/cx: gates = m@Wih^T + (bih+bhh)
        gemm(m_bf, mWih_bf, nullptr, nullptr, bias_m, gates,
             Bb, Gp, Mpad, Mpad, Mpad, Gp);
        k_lstm_pointwise<<<(nBH + 255) / 256, 256, 0, stream>>>(gates, zero_f32, c_dummy,
                                                                hm_f32, hm_bf, Bb, H, Hpad, Gp);
    }

    // --- final FC: [256 x 300] @ [3 x 300]^T + b ---
    k_fc<<<(Bb * Cc + 7) / 8, 256, 0, stream>>>(hm_f32, fc_w, fc_b, out, Bb, H, Cc);
}
